// ImportanceRenderer_77670188581439
// MI455X (gfx1250) — matrix operand order repaired and checked by reading, not compiled, not run on hardware
//
#include <hip/hip_runtime.h>
#include <hip/hip_bf16.h>
#include <math.h>

#define B_   4
#define R_   4096
#define S_   48
#define NI_  48
#define C_   32
#define HW_  256
#define HID_ 64

typedef __attribute__((ext_vector_type(16))) _Float16 v16h;
typedef __attribute__((ext_vector_type(8)))  _Float16 v8h;
typedef __attribute__((ext_vector_type(8)))  float    v8f;
typedef __attribute__((ext_vector_type(4)))  unsigned int v4u;
typedef __attribute__((ext_vector_type(8)))  int      v8i_;
typedef __attribute__((ext_vector_type(4)))  int      v4i_;

// ---- raw HW transcendentals (v_exp_f32 / v_log_f32 / v_rcp_f32), guarded ----
__device__ __forceinline__ float fast_exp(float x) {
#if __has_builtin(__builtin_amdgcn_exp2f)
    return __builtin_amdgcn_exp2f(x * 1.44269504f);   // e^x = 2^(x*log2(e))
#else
    return __expf(x);
#endif
}
__device__ __forceinline__ float fast_log(float x) {
#if __has_builtin(__builtin_amdgcn_logf)
    return 0.69314718f * __builtin_amdgcn_logf(x);    // ln(x) = log2(x)*ln2
#else
    return __logf(x);
#endif
}
__device__ __forceinline__ float fast_rcp(float x) {
#if __has_builtin(__builtin_amdgcn_rcpf)
    return __builtin_amdgcn_rcpf(x);
#else
    return __fdividef(1.f, x);
#endif
}
__device__ __forceinline__ float softplusf(float x) {
    // stable log(1+exp(x)); pre-activations here are small, raw HW trans is safe
    return fmaxf(x, 0.f) + fast_log(1.f + fast_exp(-fabsf(x)));
}
__device__ __forceinline__ float sigmoidf_(float x) {
    return fast_rcp(1.f + fast_exp(-x));
}
__device__ __forceinline__ int clampi(int v, int lo, int hi) {
    return v < lo ? lo : (v > hi ? hi : v);
}

// ---------------------------------------------------------------------------
// Pass kernel: tri-plane bilinear features -> WMMA layer1 -> VALU layer2.
// 128 threads = 4 waves; each wave owns 32 samples (two 16x32 A tiles).
// ---------------------------------------------------------------------------
__global__ __launch_bounds__(128, 1) void tri_mlp_pass(
    const float* __restrict__ planes,   // [B][3][C][HW][HW]
    const float* __restrict__ orig,     // [B*R][3]
    const float* __restrict__ dirs,     // [B*R][3]
    const float* __restrict__ w1,       // [C][HID]
    const float* __restrict__ b1,       // [HID]
    const float* __restrict__ w2,       // [HID][4]
    const float* __restrict__ b2,       // [4]
    const float* __restrict__ depth_in, // [B*R*S] or unused
    float*       __restrict__ out_rgbs, // [B*R*S][4] = r,g,b,sigma
    int use_depth_buf)
{
    __shared__ __align__(16) float    sW1raw[C_ * HID_];   // staged f32 w1 (TDM target)
    __shared__ __align__(16) _Float16 sW1t[HID_ * C_];     // transposed f16: [n][k]
    __shared__ float    sB1[HID_];
    __shared__ float    sW2[HID_ * 4];
    __shared__ float    sB2[4];
    __shared__ __align__(16) _Float16 sFeat[4][32][C_];    // [wave][sample][k]
    __shared__ __align__(16) float    sH[4][32][HID_];     // [wave][sample][n]

    const int tid  = threadIdx.x;
    const int lane = tid & 31;
    const int wv   = tid >> 5;
    const int gid  = blockIdx.x * 128 + tid; // global sample id, exact fit

#if defined(__has_builtin) && __has_builtin(__builtin_amdgcn_tensor_load_to_lds)
    // ---- Tensor Data Mover: bulk-copy w1 (8KB, contiguous) into LDS ----
    if (wv == 0) {
        const unsigned long long gaddr = (unsigned long long)w1;
        const unsigned int ldsOff = (unsigned int)(size_t)(void*)sW1raw; // LDS byte offset
        v4u g0;
        g0[0] = 1u;                                     // count=1, is_restore=0
        g0[1] = ldsOff;                                 // lds_addr (bytes)
        g0[2] = (unsigned int)(gaddr & 0xFFFFFFFFull);  // global_addr[31:0]
        g0[3] = (unsigned int)((gaddr >> 32) & 0x01FFFFFFull) | (2u << 30); // addr[56:32] | type=2
        v8i_ g1;
        const unsigned int nelem = C_ * HID_;           // 2048 x 4B
        g1[0] = (int)(2u << 16);                        // wg_mask=0 | data_size=2 (4B)
        g1[1] = 0;                                      // atomic barrier addr = 0
        g1[2] = (int)((nelem & 0xFFFFu) << 16);         // tensor_dim0 low bits (bits 48..63)
        g1[3] = (int)((nelem >> 16) | (1u << 16));      // tensor_dim0 hi | tensor_dim1=1
        g1[4] = (int)((0u) | (nelem << 16));            // tensor_dim1 hi=0 | tile_dim0
        g1[5] = 1;                                      // tile_dim1=1 | tile_dim2=0
        g1[6] = (int)nelem;                             // tensor_dim0_stride low 32
        g1[7] = 0;                                      // stride0 hi | stride1 = 0
        v4i_ z4 = {0, 0, 0, 0};
#if __clang_major__ >= 23
        v8i_ z8 = {0, 0, 0, 0, 0, 0, 0, 0};
        __builtin_amdgcn_tensor_load_to_lds(g0, g1, z4, z4, z8, 0);
#else
        __builtin_amdgcn_tensor_load_to_lds(g0, g1, z4, z4, 0);
#endif
        __builtin_amdgcn_s_wait_tensorcnt(0);
    }
#else
    for (int i = tid; i < C_ * HID_; i += 128) sW1raw[i] = w1[i];
#endif
    // stage small tensors (plain)
    for (int i = tid; i < HID_;     i += 128) sB1[i] = b1[i];
    for (int i = tid; i < HID_ * 4; i += 128) sW2[i] = w2[i];
    if (tid < 4) sB2[tid] = b2[tid];
    __syncthreads();   // sW1raw (TDM) + small tensors visible block-wide

    // transpose+convert w1 -> f16 [n][k] so B fragments are contiguous
    for (int i = tid; i < HID_ * C_; i += 128) {
        int k = i & (C_ - 1), n = i >> 5;
        sW1t[i] = (_Float16)sW1raw[k * HID_ + n];
    }

    // ---------------- per-sample tri-plane feature gather ----------------
    const int si  = gid % S_;
    const int ray = gid / S_;
    const int b   = ray / R_;

    const float ox = orig[ray * 3 + 0], oy = orig[ray * 3 + 1], oz = orig[ray * 3 + 2];
    const float dx = dirs[ray * 3 + 0], dy = dirs[ray * 3 + 1], dz = dirs[ray * 3 + 2];

    float zdepth;
    if (use_depth_buf) {
        zdepth = depth_in[gid];
    } else {
        float cam = sqrtf(ox * ox + oy * oy + oz * oz);
        zdepth = (cam - 0.5f) + ((float)si + 0.5f) * (1.0f / (float)S_); // far-near == 1
    }
    const float px = ox + zdepth * dx;
    const float py = oy + zdepth * dy;
    const float pz = oz + zdepth * dz;

    // plane projections: P0:(x,y)  P1:(z,x)  P2:(y,z)
    const float uu[3] = { px, pz, py };
    const float vv[3] = { py, px, pz };

    int   off[3][4];
    float wt [3][4];
#pragma unroll
    for (int p = 0; p < 3; ++p) {
        float u = uu[p] * 128.f + 127.5f;   // (c+1)*0.5*256 - 0.5
        float v = vv[p] * 128.f + 127.5f;
        float fx0 = floorf(u), fy0 = floorf(v);
        float fx = u - fx0, fy = v - fy0;
        int x0 = (int)fx0, y0 = (int)fy0;
        int   xs[2] = { x0, x0 + 1 }, ys[2] = { y0, y0 + 1 };
        float wx[2] = { 1.f - fx, fx }, wy[2] = { 1.f - fy, fy };
#pragma unroll
        for (int ty = 0; ty < 2; ++ty)
#pragma unroll
            for (int tx = 0; tx < 2; ++tx) {
                int xi = xs[tx], yi = ys[ty];
                bool valid = (xi >= 0) && (xi < HW_) && (yi >= 0) && (yi < HW_);
                off[p][ty * 2 + tx] = clampi(yi, 0, HW_ - 1) * HW_ + clampi(xi, 0, HW_ - 1);
                wt [p][ty * 2 + tx] = valid ? wx[tx] * wy[ty] : 0.f;
            }
        __builtin_prefetch(planes + ((size_t)(b * 3 + p) * C_) * (HW_ * HW_) + off[p][0], 0, 1);
    }

    for (int c = 0; c < C_; ++c) {
        float acc = 0.f;
#pragma unroll
        for (int p = 0; p < 3; ++p) {
            const float* img = planes + ((size_t)(b * 3 + p) * C_ + c) * (HW_ * HW_);
            acc += img[off[p][0]] * wt[p][0] + img[off[p][1]] * wt[p][1]
                 + img[off[p][2]] * wt[p][2] + img[off[p][3]] * wt[p][3];
        }
        sFeat[wv][lane][c] = (_Float16)(acc * (1.f / 3.f)); // mean over 3 planes
    }
    __syncthreads();

    // ---------------- layer1: x(16x32) @ w1(32x64) via WMMA ----------------
    const bool hi  = lane >= 16;
    const int  l16 = lane & 15;
    const int  k8  = hi ? 8 : 0;   // A: lane-group K base
    const int  k16 = hi ? 16 : 0;  // unused; B also uses k8

    // B fragments: halves 0-7 -> K = 8*hi + j ; halves 8-15 -> K = 16 + 8*hi + j ; col N = t*16+l16
    v16h bfrag[4];
#pragma unroll
    for (int t = 0; t < 4; ++t) {
        const _Float16* brow = &sW1t[(t * 16 + l16) * C_ + k8];
        v8h b0 = *(const v8h*)(brow);
        v8h b1v = *(const v8h*)(brow + 16);
#pragma unroll
        for (int j = 0; j < 8; ++j) { bfrag[t][j] = b0[j]; bfrag[t][j + 8] = b1v[j]; }
    }

    // A fragments: halves 0-7 -> K = 8*hi + j ; halves 8-15 -> K = 16 + 8*hi + j
    v16h afrag[2];
#pragma unroll
    for (int m = 0; m < 2; ++m) {
        const _Float16* arow = &sFeat[wv][m * 16 + l16][0];
        v8h a0 = *(const v8h*)(arow + k8);
        v8h a1 = *(const v8h*)(arow + 16 + k8);
#pragma unroll
        for (int j = 0; j < 8; ++j) { afrag[m][j] = a0[j]; afrag[m][j + 8] = a1[j]; }
    }

#pragma unroll
    for (int m = 0; m < 2; ++m)
#pragma unroll
        for (int t = 0; t < 4; ++t) {
            v8f cz = {};
            v8f d = __builtin_amdgcn_wmma_f32_16x16x32_f16(
                        false, afrag[m], false, bfrag[t], (short)0, cz, false, false);
            // D layout: VGPR v -> row M = v + 8*hi, col N = lane&15
            const int n = t * 16 + l16;
            float* hrow = &sH[wv][m * 16 + (hi ? 8 : 0)][n];
#pragma unroll
            for (int v = 0; v < 8; ++v)
                hrow[(size_t)v * HID_] = softplusf(d[v] + sB1[n]);
        }
    __syncthreads();

    // ---------------- layer2: h(64) @ w2(64x4), per lane = one sample ------
    float o0 = sB2[0], o1 = sB2[1], o2 = sB2[2], o3 = sB2[3];
#pragma unroll
    for (int k4 = 0; k4 < HID_ / 4; ++k4) {
        float4 hv = *(const float4*)&sH[wv][lane][k4 * 4];
        const float* wrow = &sW2[k4 * 16];
        o0 = fmaf(hv.x, wrow[0],  o0); o1 = fmaf(hv.x, wrow[1],  o1);
        o2 = fmaf(hv.x, wrow[2],  o2); o3 = fmaf(hv.x, wrow[3],  o3);
        o0 = fmaf(hv.y, wrow[4],  o0); o1 = fmaf(hv.y, wrow[5],  o1);
        o2 = fmaf(hv.y, wrow[6],  o2); o3 = fmaf(hv.y, wrow[7],  o3);
        o0 = fmaf(hv.z, wrow[8],  o0); o1 = fmaf(hv.z, wrow[9],  o1);
        o2 = fmaf(hv.z, wrow[10], o2); o3 = fmaf(hv.z, wrow[11], o3);
        o0 = fmaf(hv.w, wrow[12], o0); o1 = fmaf(hv.w, wrow[13], o1);
        o2 = fmaf(hv.w, wrow[14], o2); o3 = fmaf(hv.w, wrow[15], o3);
    }
    float sigma = 10.f * softplusf(-10.f * o0);
    float4 outv;
    outv.x = sigmoidf_(o1) * 1.002f - 0.001f;
    outv.y = sigmoidf_(o2) * 1.002f - 0.001f;
    outv.z = sigmoidf_(o3) * 1.002f - 0.001f;
    outv.w = sigma;
    ((float4*)out_rgbs)[gid] = outv;
}

// ---------------------------------------------------------------------------
// Coarse ray march -> max-pooled weights -> inverse-CDF importance samples.
// One thread per ray.
// ---------------------------------------------------------------------------
__global__ __launch_bounds__(256, 1) void importance_kernel(
    const float* __restrict__ orig,
    const float* __restrict__ rgbs_c,   // [B*R*S][4]
    float*       __restrict__ depths_f) // [B*R][NI]
{
    const int ray = blockIdx.x * 256 + threadIdx.x;
    const float ox = orig[ray * 3 + 0], oy = orig[ray * 3 + 1], oz = orig[ray * 3 + 2];
    const float near = sqrtf(ox * ox + oy * oy + oz * oz) - 0.5f;
    const float delta = 1.f / (float)S_;

    float sig[S_];
    for (int i = 0; i < S_; ++i)
        sig[i] = rgbs_c[((size_t)ray * S_ + i) * 4 + 3];

    float wgt[S_ - 1];
    float T = 1.f;
    for (int i = 0; i < S_ - 1; ++i) {
        float a = 1.f - fast_exp(-delta * 0.5f * (sig[i] + sig[i + 1]));
        wgt[i] = a * T;
        T *= (1.f - a + 1e-10f);
    }
    // wm = max of zero-padded neighbors, then 0.5*(wm+wm_next)+0.01
    float wmx[S_];
    for (int i = 0; i < S_; ++i) {
        float l = (i == 0) ? 0.f : wgt[i - 1];
        float r = (i == S_ - 1) ? 0.f : wgt[i];
        wmx[i] = fmaxf(l, r);
    }
    float wsm[S_ - 1];
    for (int i = 0; i < S_ - 1; ++i)
        wsm[i] = 0.5f * (wmx[i] + wmx[i + 1]) + 0.01f;

    // pdf over wsm[1..45] (45 bins), cdf[0..45] with leading zero
    float cdf[S_ - 2];
    cdf[0] = 0.f;
    float tot = 0.f;
    for (int k = 0; k < S_ - 3; ++k) tot += wsm[k + 1] + 1e-5f;
    float invTot = fast_rcp(tot);
    float acc = 0.f;
    for (int k = 0; k < S_ - 3; ++k) {
        acc += (wsm[k + 1] + 1e-5f) * invTot;
        cdf[k + 1] = acc;
    }
    for (int k = 0; k < NI_; ++k) {
        float u = (float)k * (1.f / (float)(NI_ - 1));
        int ind = 0;
        while (ind < S_ - 2 && cdf[ind] <= u) ++ind;   // searchsorted 'right'
        int below = ind - 1; if (below < 0) below = 0;
        int above = ind;     if (above > S_ - 3) above = S_ - 3;
        float cb = cdf[below], ca = cdf[above];
        float zb = near + (float)(below + 1) * delta;  // z_mid[i] = near+(i+1)/S
        float za = near + (float)(above + 1) * delta;
        float den = (ca - cb < 1e-5f) ? 1.f : (ca - cb);
        depths_f[(size_t)ray * NI_ + k] = zb + (u - cb) * fast_rcp(den) * (za - zb);
    }
}

// ---------------------------------------------------------------------------
// Streaming stable merge (coarse wins ties) + final ray march.
// One thread per ray.
// ---------------------------------------------------------------------------
__global__ __launch_bounds__(256, 1) void final_march_kernel(
    const float* __restrict__ orig,
    const float* __restrict__ rgbs_c,
    const float* __restrict__ rgbs_f,
    const float* __restrict__ depths_f,
    float*       __restrict__ out)
{
    const int ray = blockIdx.x * 256 + threadIdx.x;
    const float ox = orig[ray * 3 + 0], oy = orig[ray * 3 + 1], oz = orig[ray * 3 + 2];
    const float near = sqrtf(ox * ox + oy * oy + oz * oz) - 0.5f;
    const float delta = 1.f / (float)S_;

    int i = 0, j = 0;
    float dc = near + 0.5f * delta;
    float df = depths_f[(size_t)ray * NI_];

    float pd, pr, pg, pb, ps;
    auto pick = [&](float& d_, float& r_, float& g_, float& b_, float& s_) {
        bool takeC = (i < S_) && (j >= NI_ || dc <= df); // stable: coarse first on tie
        if (takeC) {
            float4 v = ((const float4*)rgbs_c)[(size_t)ray * S_ + i];
            d_ = dc; r_ = v.x; g_ = v.y; b_ = v.z; s_ = v.w;
            ++i; dc = near + ((float)i + 0.5f) * delta;
        } else {
            float4 v = ((const float4*)rgbs_f)[(size_t)ray * NI_ + j];
            d_ = df; r_ = v.x; g_ = v.y; b_ = v.z; s_ = v.w;
            ++j; df = (j < NI_) ? depths_f[(size_t)ray * NI_ + j] : 3.4e38f;
        }
    };

    pick(pd, pr, pg, pb, ps);
    float T = 1.f, aR = 0.f, aG = 0.f, aB = 0.f, aD = 0.f, aW = 0.f;
    for (int s = 1; s < S_ + NI_; ++s) {
        float cd, cr, cg, cb, cs;
        pick(cd, cr, cg, cb, cs);
        float dl = cd - pd;
        float a = 1.f - fast_exp(-dl * 0.5f * (ps + cs));
        float w = a * T;
        T *= (1.f - a + 1e-10f);
        aR += w * 0.5f * (pr + cr);
        aG += w * 0.5f * (pg + cg);
        aB += w * 0.5f * (pb + cb);
        aD += w * 0.5f * (pd + cd);
        aW += w;
        pd = cd; pr = cr; pg = cg; pb = cb; ps = cs;
    }

    const int BR = B_ * R_;
    out[(size_t)ray * 3 + 0] = aR;
    out[(size_t)ray * 3 + 1] = aG;
    out[(size_t)ray * 3 + 2] = aB;
    out[(size_t)BR * 3 + ray] = aD;
    out[(size_t)BR * 4 + ray] = aW;
}

// ---------------------------------------------------------------------------
extern "C" void kernel_launch(void* const* d_in, const int* in_sizes, int n_in,
                              void* d_out, int out_size, void* d_ws, size_t ws_size,
                              hipStream_t stream) {
    (void)in_sizes; (void)n_in; (void)out_size; (void)ws_size;
    const float* planes = (const float*)d_in[0];
    const float* orig   = (const float*)d_in[1];
    const float* dirs   = (const float*)d_in[2];
    const float* w1     = (const float*)d_in[3];
    const float* b1     = (const float*)d_in[4];
    const float* w2     = (const float*)d_in[5];
    const float* b2     = (const float*)d_in[6];
    float* out = (float*)d_out;

    char* ws = (char*)d_ws;
    const size_t passBytes = (size_t)B_ * R_ * S_ * 4 * sizeof(float); // 12.6 MB
    float* rgbs_c  = (float*)(ws);
    float* rgbs_f  = (float*)(ws + passBytes);
    float* depthsF = (float*)(ws + 2 * passBytes);

    const int nSamp = B_ * R_ * S_;             // 786432
    dim3 gPass(nSamp / 128), bPass(128);
    dim3 gRay((B_ * R_) / 256), bRay(256);

    tri_mlp_pass<<<gPass, bPass, 0, stream>>>(planes, orig, dirs, w1, b1, w2, b2,
                                              depthsF /*unused*/, rgbs_c, 0);
    importance_kernel<<<gRay, bRay, 0, stream>>>(orig, rgbs_c, depthsF);
    tri_mlp_pass<<<gPass, bPass, 0, stream>>>(planes, orig, dirs, w1, b1, w2, b2,
                                              depthsF, rgbs_f, 1);
    final_march_kernel<<<gRay, bRay, 0, stream>>>(orig, rgbs_c, rgbs_f, depthsF, out);
}
